// TreeHopNode_85280870629918
// MI455X (gfx1250) — matrix operand order repaired and checked by reading, not compiled, not run on hardware
//
#include <hip/hip_runtime.h>
#include <hip/hip_bf16.h>

// ---------------------------------------------------------------------------
// MI455X (gfx1250): bf16 WMMA pipeline, f32 accumulate. Ping-pong double
// buffering (manual 2x unroll, two named buffer sets) so fragment loads
// overlap WMMAs with no register copies and no spills.
// ---------------------------------------------------------------------------

typedef __bf16 bf16;
typedef __attribute__((ext_vector_type(8)))  __bf16 v8bf;
typedef __attribute__((ext_vector_type(16))) __bf16 v16bf;
typedef __attribute__((ext_vector_type(8)))  float  v8f;

#define NROWS 65536
#define EMBED 1024
#define GDIM  256
#define MLPD  256
#define NHEAD 4
#define HG    1024   // NHEAD * GDIM

// A-fragment (16x32 bf16, ISA 7.12.2): per lane two 16B chunks at +0 / +16
// elements (caller pre-adds the lane-dependent 0/8 K sub-offset).
static __device__ __forceinline__ v16bf load_a16(const bf16* p) {
  v8bf lo = *(const v8bf*)(p);
  v8bf hi = *(const v8bf*)(p + 16);
  v16bf r;
#pragma unroll
  for (int i = 0; i < 8; ++i) { r[i] = lo[i]; r[i + 8] = hi[i]; }
  return r;
}
// B-fragment (32x16 bf16) from pre-transposed weights: lane n reads two 16B
// chunks of column n at +0 / +16 (caller pre-adds the 0/8 K sub-offset).
static __device__ __forceinline__ v16bf load_b16(const bf16* p) {
  v8bf lo = *(const v8bf*)(p);
  v8bf hi = *(const v8bf*)(p + 16);
  v16bf r;
#pragma unroll
  for (int i = 0; i < 8; ++i) { r[i] = lo[i]; r[i + 8] = hi[i]; }
  return r;
}

#define WMMA_BF16(a, b, c) \
  __builtin_amdgcn_wmma_f32_16x16x32_bf16(false, (a), false, (b), (short)0, (c), false, false)

// --------------------------- conversion kernels ----------------------------

__global__ __launch_bounds__(256) void k_cvt_act(
    const float4* __restrict__ q4, const float4* __restrict__ rep4,
    v8bf* __restrict__ qb8, v8bf* __restrict__ rb8, int total8) {
  int stride = gridDim.x * blockDim.x;
  for (int i = blockIdx.x * blockDim.x + threadIdx.x; i < total8; i += stride) {
    float4 a0 = q4[2 * i], a1 = q4[2 * i + 1];
    float4 b0 = rep4[2 * i], b1 = rep4[2 * i + 1];
    v8bf qa, ra;
    qa[0] = (bf16)a0.x; qa[1] = (bf16)a0.y; qa[2] = (bf16)a0.z; qa[3] = (bf16)a0.w;
    qa[4] = (bf16)a1.x; qa[5] = (bf16)a1.y; qa[6] = (bf16)a1.z; qa[7] = (bf16)a1.w;
    ra[0] = (bf16)b0.x; ra[1] = (bf16)b0.y; ra[2] = (bf16)b0.z; ra[3] = (bf16)b0.w;
    ra[4] = (bf16)b1.x; ra[5] = (bf16)b1.y; ra[6] = (bf16)b1.z; ra[7] = (bf16)b1.w;
    qb8[i] = qa;
    rb8[i] = ra;
  }
}

// src: [batch][R][C] f32 row-major  ->  dst: [batch][C][R] bf16 row-major
__global__ __launch_bounds__(256) void k_tconv(
    const float* __restrict__ src, bf16* __restrict__ dst,
    int R, int C, int total) {
  int stride = gridDim.x * blockDim.x;
  for (int i = blockIdx.x * blockDim.x + threadIdx.x; i < total; i += stride) {
    int r = i % R;
    int t = i / R;
    int c = t % C;
    int b = t / C;
    dst[(size_t)b * C * R + (size_t)c * R + r] =
        (bf16)src[(size_t)b * R * C + (size_t)r * C + c];
  }
}

// ------------------- fused Q/K/V projection + sigmoid gate -----------------
// z = sigmoid((q@Wq)*(rep@Wk)/sqrt(G)) * (rep@Wv), written as bf16 [N, HG].

struct Frag5 { v16bf aq, ar, fq, fk, fv; };

static __device__ __forceinline__ Frag5 load5(
    const bf16* qrow, const bf16* rrow,
    const bf16* bq, const bf16* bk, const bf16* bv, int k) {
  Frag5 f;
  f.aq = load_a16(qrow + k);
  f.ar = load_a16(rrow + k);
  f.fq = load_b16(bq + k);
  f.fk = load_b16(bk + k);
  f.fv = load_b16(bv + k);
  return f;
}

__global__ __launch_bounds__(256)
__attribute__((amdgpu_waves_per_eu(1)))
void k_qkvz(
    const bf16* __restrict__ qb, const bf16* __restrict__ rb,
    const bf16* __restrict__ wqT, const bf16* __restrict__ wkT,
    const bf16* __restrict__ wvT, bf16* __restrict__ zb) {
  const int lane = threadIdx.x & 31;
  const int wave = threadIdx.x >> 5;
  const int ln   = lane & 15;
  const int hiH  = lane >> 4;
  const int row0 = blockIdx.x * 128 + wave * 16;

  const bf16* qrow = qb + (size_t)(row0 + ln) * EMBED + (hiH ? 8 : 0);
  const bf16* rrow = rb + (size_t)(row0 + ln) * EMBED + (hiH ? 8 : 0);
  const int   bOff = hiH ? 8 : 0;

  // Warm WGP$/L2 with this wave's A rows (global_prefetch_b8).
#pragma unroll
  for (int pk = 0; pk < EMBED; pk += 256) {
    __builtin_prefetch(qrow + pk, 0, 3);
    __builtin_prefetch(rrow + pk, 0, 3);
  }

  for (int n0 = 0; n0 < HG; n0 += 16) {
    const bf16* bq = wqT + (size_t)(n0 + ln) * EMBED + bOff;
    const bf16* bk = wkT + (size_t)(n0 + ln) * EMBED + bOff;
    const bf16* bv = wvT + (size_t)(n0 + ln) * EMBED + bOff;
    v8f accQ = {}, accK = {}, accV = {};

    Frag5 s0 = load5(qrow, rrow, bq, bk, bv, 0);
    Frag5 s1;
#pragma unroll 1
    for (int k0 = 0; k0 < EMBED - 64; k0 += 64) {
      s1 = load5(qrow, rrow, bq, bk, bv, k0 + 32);
      accQ = WMMA_BF16(s0.aq, s0.fq, accQ);
      accK = WMMA_BF16(s0.ar, s0.fk, accK);
      accV = WMMA_BF16(s0.ar, s0.fv, accV);
      s0 = load5(qrow, rrow, bq, bk, bv, k0 + 64);
      accQ = WMMA_BF16(s1.aq, s1.fq, accQ);
      accK = WMMA_BF16(s1.ar, s1.fk, accK);
      accV = WMMA_BF16(s1.ar, s1.fv, accV);
    }
    // epilogue: k = EMBED-64 (s0) and k = EMBED-32 (s1)
    s1 = load5(qrow, rrow, bq, bk, bv, EMBED - 32);
    accQ = WMMA_BF16(s0.aq, s0.fq, accQ);
    accK = WMMA_BF16(s0.ar, s0.fk, accK);
    accV = WMMA_BF16(s0.ar, s0.fv, accV);
    accQ = WMMA_BF16(s1.aq, s1.fq, accQ);
    accK = WMMA_BF16(s1.ar, s1.fk, accK);
    accV = WMMA_BF16(s1.ar, s1.fv, accV);

#pragma unroll
    for (int r = 0; r < 8; ++r) {
      float g = accQ[r] * accK[r] * 0.0625f;      // 1/sqrt(256)
      float s = 1.0f / (1.0f + __expf(-g));
      int row = row0 + (hiH ? 8 + r : r);
      zb[(size_t)row * HG + n0 + ln] = (bf16)(s * accV[r]);
    }
  }
}

// ------------------------- per-head MLP (G->MLP->G) ------------------------
// Reads z, writes update_gate in-place over z. h1 staged through LDS so it
// can be re-fed as a WMMA A-operand.

__global__ __launch_bounds__(256)
__attribute__((amdgpu_waves_per_eu(1)))
void k_mlp(
    const bf16* __restrict__ w1T, const bf16* __restrict__ w2T,
    const float* __restrict__ b1, const float* __restrict__ b2,
    bf16* __restrict__ zb) {
  __shared__ bf16 h1s[8][16][264];   // per-wave 16x256 strip, padded stride
  const int lane = threadIdx.x & 31;
  const int wave = threadIdx.x >> 5;
  const int ln   = lane & 15;
  const int hiH  = lane >> 4;
  const int row0 = blockIdx.x * 128 + wave * 16;
  const int aOff = hiH ? 8 : 0;
  const int bOff = hiH ? 8 : 0;

  for (int h = 0; h < NHEAD; ++h) {
    const bf16* arow = zb + (size_t)(row0 + ln) * HG + h * GDIM + aOff;

    // layer 1: h1 = relu(z @ W1[h] + b1[h])  -> LDS
    for (int n0 = 0; n0 < MLPD; n0 += 16) {
      const bf16* bw = w1T + ((size_t)h * MLPD + n0 + ln) * GDIM + bOff;
      v8f acc = {};
      v16bf a0 = load_a16(arow), b0 = load_b16(bw);
      v16bf a1, b1f;
#pragma unroll 1
      for (int k0 = 0; k0 < GDIM - 64; k0 += 64) {
        a1 = load_a16(arow + k0 + 32); b1f = load_b16(bw + k0 + 32);
        acc = WMMA_BF16(a0, b0, acc);
        a0 = load_a16(arow + k0 + 64); b0 = load_b16(bw + k0 + 64);
        acc = WMMA_BF16(a1, b1f, acc);
      }
      a1 = load_a16(arow + GDIM - 32); b1f = load_b16(bw + GDIM - 32);
      acc = WMMA_BF16(a0, b0, acc);
      acc = WMMA_BF16(a1, b1f, acc);

      float bias = b1[h * MLPD + n0 + ln];
#pragma unroll
      for (int r = 0; r < 8; ++r) {
        float v = acc[r] + bias;
        h1s[wave][hiH ? 8 + r : r][n0 + ln] = (bf16)(v > 0.0f ? v : 0.0f);
      }
    }
    __syncthreads();

    // layer 2: ug = h1 @ W2[h] + b2[h]  -> overwrite z columns of this head
    const bf16* a2 = &h1s[wave][ln][aOff];
    for (int n0 = 0; n0 < GDIM; n0 += 16) {
      const bf16* bw = w2T + ((size_t)h * GDIM + n0 + ln) * MLPD + bOff;
      v8f acc = {};
      v16bf a0 = load_a16(a2), b0 = load_b16(bw);
      v16bf a1, b1f;
#pragma unroll 1
      for (int k0 = 0; k0 < MLPD - 64; k0 += 64) {
        a1 = load_a16(a2 + k0 + 32); b1f = load_b16(bw + k0 + 32);
        acc = WMMA_BF16(a0, b0, acc);
        a0 = load_a16(a2 + k0 + 64); b0 = load_b16(bw + k0 + 64);
        acc = WMMA_BF16(a1, b1f, acc);
      }
      a1 = load_a16(a2 + MLPD - 32); b1f = load_b16(bw + MLPD - 32);
      acc = WMMA_BF16(a0, b0, acc);
      acc = WMMA_BF16(a1, b1f, acc);

      float bias = b2[h * GDIM + n0 + ln];
#pragma unroll
      for (int r = 0; r < 8; ++r) {
        int row = row0 + (hiH ? 8 + r : r);
        zb[(size_t)row * HG + h * GDIM + n0 + ln] = (bf16)(acc[r] + bias);
      }
    }
    __syncthreads();
  }
}

// ------------------- final projection with fused residual ------------------
// out = q - rep + ug @ Ws.  Two N-tiles per pass: each A-fragment feeds two
// WMMAs (halves A traffic, doubles independent WMMA work between waits).

struct Frag3 { v16bf a, b0, b1; };

static __device__ __forceinline__ Frag3 load3(
    const bf16* arow, const bf16* bw0, const bf16* bw1, int k) {
  Frag3 f;
  f.a  = load_a16(arow + k);
  f.b0 = load_b16(bw0 + k);
  f.b1 = load_b16(bw1 + k);
  return f;
}

__global__ __launch_bounds__(256)
__attribute__((amdgpu_waves_per_eu(1)))
void k_final(
    const bf16* __restrict__ ug, const bf16* __restrict__ wsT,
    const float* __restrict__ q, const float* __restrict__ rep,
    float* __restrict__ out) {
  const int lane = threadIdx.x & 31;
  const int wave = threadIdx.x >> 5;
  const int ln   = lane & 15;
  const int hiH  = lane >> 4;
  const int row0 = blockIdx.x * 128 + wave * 16;

  const bf16* arow = ug + (size_t)(row0 + ln) * HG + (hiH ? 8 : 0);
  const int   bOff = hiH ? 8 : 0;

#pragma unroll
  for (int pk = 0; pk < HG; pk += 256) __builtin_prefetch(arow + pk, 0, 3);

  for (int n0 = 0; n0 < EMBED; n0 += 32) {
    const bf16* bw0 = wsT + (size_t)(n0 + ln) * HG + bOff;
    const bf16* bw1 = wsT + (size_t)(n0 + 16 + ln) * HG + bOff;
    v8f acc0 = {}, acc1 = {};

    Frag3 s0 = load3(arow, bw0, bw1, 0);
    Frag3 s1;
#pragma unroll 1
    for (int k0 = 0; k0 < HG - 64; k0 += 64) {
      s1 = load3(arow, bw0, bw1, k0 + 32);
      acc0 = WMMA_BF16(s0.a, s0.b0, acc0);
      acc1 = WMMA_BF16(s0.a, s0.b1, acc1);
      s0 = load3(arow, bw0, bw1, k0 + 64);
      acc0 = WMMA_BF16(s1.a, s1.b0, acc0);
      acc1 = WMMA_BF16(s1.a, s1.b1, acc1);
    }
    s1 = load3(arow, bw0, bw1, HG - 32);
    acc0 = WMMA_BF16(s0.a, s0.b0, acc0);
    acc1 = WMMA_BF16(s0.a, s0.b1, acc1);
    acc0 = WMMA_BF16(s1.a, s1.b0, acc0);
    acc1 = WMMA_BF16(s1.a, s1.b1, acc1);

#pragma unroll
    for (int r = 0; r < 8; ++r) {
      size_t row = (size_t)(row0 + (hiH ? 8 + r : r));
      size_t i0 = row * EMBED + n0 + ln;
      size_t i1 = i0 + 16;
      out[i0] = acc0[r] + q[i0] - rep[i0];
      out[i1] = acc1[r] + q[i1] - rep[i1];
    }
  }
}

// --------------------------------- launch ----------------------------------

extern "C" void kernel_launch(void* const* d_in, const int* in_sizes, int n_in,
                              void* d_out, int out_size, void* d_ws, size_t ws_size,
                              hipStream_t stream) {
  const float* q   = (const float*)d_in[0];
  const float* rep = (const float*)d_in[1];
  const float* Wq  = (const float*)d_in[2];
  const float* Wk  = (const float*)d_in[3];
  const float* Wv  = (const float*)d_in[4];
  const float* W1  = (const float*)d_in[5];
  const float* b1  = (const float*)d_in[6];
  const float* W2  = (const float*)d_in[7];
  const float* b2  = (const float*)d_in[8];
  const float* Ws  = (const float*)d_in[9];
  float* out = (float*)d_out;

  // Workspace carve-up (bf16 buffers), 256B-aligned by construction.
  char* ws = (char*)d_ws;
  const size_t NB = (size_t)NROWS * EMBED;          // 67,108,864 elems
  bf16* qb  = (bf16*)(ws);                          // NB
  bf16* rb  = qb + NB;                              // NB
  bf16* zb  = rb + NB;                              // NB (z, then ug in-place)
  bf16* wqT = zb + NB;                              // EMBED*HG
  bf16* wkT = wqT + (size_t)EMBED * HG;
  bf16* wvT = wkT + (size_t)EMBED * HG;
  bf16* w1T = wvT + (size_t)EMBED * HG;             // NHEAD*MLPD*GDIM
  bf16* w2T = w1T + (size_t)NHEAD * MLPD * GDIM;    // NHEAD*GDIM*MLPD
  bf16* wsT = w2T + (size_t)NHEAD * GDIM * MLPD;    // EMBED*HG

  // 1) activations -> bf16 (vectorized: float4 x2 in, 16B bf16 out)
  k_cvt_act<<<8192, 256, 0, stream>>>((const float4*)q, (const float4*)rep,
                                      (v8bf*)qb, (v8bf*)rb, (int)(NB / 8));

  // 2) weights -> transposed bf16 (B^T so B-fragments are contiguous loads)
  k_tconv<<<4096, 256, 0, stream>>>(Wq, wqT, EMBED, HG, EMBED * HG);
  k_tconv<<<4096, 256, 0, stream>>>(Wk, wkT, EMBED, HG, EMBED * HG);
  k_tconv<<<4096, 256, 0, stream>>>(Wv, wvT, EMBED, HG, EMBED * HG);
  k_tconv<<<1024, 256, 0, stream>>>(W1, w1T, GDIM, MLPD, NHEAD * GDIM * MLPD);
  k_tconv<<<1024, 256, 0, stream>>>(W2, w2T, MLPD, GDIM, NHEAD * MLPD * GDIM);
  k_tconv<<<4096, 256, 0, stream>>>(Ws, wsT, HG, EMBED, HG * EMBED);

  // 3) fused QKV + gate -> z
  k_qkvz<<<NROWS / 128, 256, 0, stream>>>(qb, rb, wqT, wkT, wvT, zb);

  // 4) per-head MLP, ug written in place over z
  k_mlp<<<NROWS / 128, 256, 0, stream>>>(w1T, w2T, b1, b2, zb);

  // 5) out = q - rep + ug @ Ws
  k_final<<<NROWS / 128, 256, 0, stream>>>(zb, wsT, q, rep, out);
}
